// Grid2MeshGNN_58171037057100
// MI455X (gfx1250) — matrix operand order repaired and checked by reading, not compiled, not run on hardware
//
#include <hip/hip_runtime.h>
#include <hip/hip_bf16.h>

// ---------------------------------------------------------------------------
// Grid2Mesh GNN for MI455X (gfx1250): fused 2-layer MLPs on WMMA bf16->f32,
// async-LDS staging, L2-resident bf16 gather table.
//   d_out = [grid_out (N_GRID x 256) | mesh_new (N_MESH x 256)] as f32.
// ---------------------------------------------------------------------------

#define DD   256        // feature dim
#define ROWS 128        // rows per block (8 row-tiles of 16); 128 KB LDS
#define NRT  (ROWS / 16)

typedef __attribute__((ext_vector_type(16))) __bf16 v16bf;
typedef __attribute__((ext_vector_type(8)))  float  v8f;

union BF16x16 { uint4 u[2]; v16bf v; };

// float -> bf16 bits, round-to-nearest-even (fallback path)
__device__ inline unsigned short f2bf(float f) {
    union { float f; unsigned u; } U;
    U.f = f;
    unsigned u = U.u;
    u += 0x7FFFu + ((u >> 16) & 1u);
    return (unsigned short)(u >> 16);
}

// two floats -> packed 2x bf16 (one v_cvt_pk_bf16_f32 when available)
#if __has_builtin(__builtin_amdgcn_cvt_pk_bf16_f32)
typedef __attribute__((ext_vector_type(2))) __bf16 v2bf;
__device__ inline unsigned pack2bf(float a, float b) {
    v2bf r = __builtin_amdgcn_cvt_pk_bf16_f32(a, b);
    union { v2bf v; unsigned u; } U;
    U.v = r;
    return U.u;
}
#else
__device__ inline unsigned pack2bf(float a, float b) {
    return (unsigned)f2bf(a) | ((unsigned)f2bf(b) << 16);
}
#endif

// relu in one instruction: v_med3_f32(v, 0, +inf)
__device__ inline float relu1(float v) {
#if __has_builtin(__builtin_amdgcn_fmed3f)
    return __builtin_amdgcn_fmed3f(v, 0.0f, __builtin_inff());
#else
    return fmaxf(v, 0.0f);
#endif
}

// A fragment (16x32 bf16) from LDS, row-major [rows][256] bf16 bits.
// Lane L: M = L&15; K = kbase..kbase+7 and kbase+16..kbase+23, kbase = 8*(L>=16).
__device__ inline v16bf lds_a_frag(const unsigned short* base, int rowtile, int k0, int lane) {
    int m  = lane & 15;
    int kb = k0 + ((lane >> 4) << 3);
    const unsigned short* p = base + (rowtile * 16 + m) * DD + kb;
    BF16x16 U;
    U.u[0] = *(const uint4*)(p);
    U.u[1] = *(const uint4*)(p + 16);
    return U.v;
}

// B fragment (32x16 bf16) from global transposed weights Wt[N][K] (bf16 bits).
// Lane L: N = L&15; K = kbase..kbase+7 and kbase+16..kbase+23, kbase = k0 + 8*(L>=16).
__device__ inline v16bf glb_b_frag(const unsigned short* Wt, int n0, int k0, int lane) {
    int n  = n0 + (lane & 15);
    int kb = k0 + ((lane >> 4) << 3);
    const unsigned short* p = Wt + n * DD + kb;
    BF16x16 U;
    U.u[0] = *(const uint4*)(p);
    U.u[1] = *(const uint4*)(p + 16);
    return U.v;
}

// One 16x16 output tile: acc = bias; acc += A(LDS) @ B(Wt) over K=256.
__device__ inline v8f tile_acc(const unsigned short* a_lds, const unsigned short* Wt,
                               const float* bvec, int rt, int col0, int lane) {
    float bias = bvec[col0 + (lane & 15)];
    v8f acc = {bias, bias, bias, bias, bias, bias, bias, bias};
#pragma unroll
    for (int k0 = 0; k0 < DD; k0 += 32) {
        v16bf av = lds_a_frag(a_lds, rt, k0, lane);
        v16bf bv = glb_b_frag(Wt, col0, k0, lane);
        acc = __builtin_amdgcn_wmma_f32_16x16x32_bf16(false, av, false, bv,
                                                      (short)0, acc, false, false);
    }
    return acc;
}

// Layer 1: hid = relu(X @ W1 + b1), X staged in LDS (bf16), hid written to LDS.
// Wave w owns columns [w*32, w*32+32).
__device__ inline void layer1(const unsigned short* xbf, unsigned short* hid,
                              const unsigned short* W1t, const float* b1,
                              int lane, int wave) {
    int n0 = wave * 32;
#pragma unroll
    for (int ct = 0; ct < 2; ++ct) {
        int col0 = n0 + ct * 16;
        for (int rt = 0; rt < NRT; ++rt) {
            v8f acc = tile_acc(xbf, W1t, b1, rt, col0, lane);
            int mbase = (lane < 16) ? 0 : 8;
            int col   = col0 + (lane & 15);
#pragma unroll
            for (int j = 0; j < 4; ++j) {
                unsigned p = pack2bf(relu1(acc[2 * j]), relu1(acc[2 * j + 1]));
                int rowa = rt * 16 + mbase + 2 * j;
                hid[rowa * DD + col]       = (unsigned short)p;
                hid[(rowa + 1) * DD + col] = (unsigned short)(p >> 16);
            }
        }
    }
}

// CDNA5 async copy: one b128 (16 B) per lane from global bf16 row data to LDS.
// LDS byte address taken from the first VGPR operand; global addr = SADDR + VADDR (GVS mode).
__device__ inline void async_row16(unsigned lds_off, unsigned g_off,
                                   const unsigned short* base) {
    asm volatile("global_load_async_to_lds_b128 %0, %1, %2"
                 :: "v"(lds_off), "v"(g_off), "s"(base) : "memory");
}
__device__ inline void async_wait0() {
    asm volatile("s_wait_asynccnt 0" ::: "memory");
}

// ---------------------------------------------------------------------------
// Prep kernels
// ---------------------------------------------------------------------------
// W (f32, [K][N] row-major) -> Wt (bf16 bits, [N][K] row-major)
__global__ __launch_bounds__(256) void transpose_bf16_kernel(const float* __restrict__ W,
                                                             unsigned short* __restrict__ Wt) {
    int idx = blockIdx.x * 256 + threadIdx.x;   // 65536 threads
    int n = idx >> 8;
    int k = idx & 255;
    Wt[n * DD + k] = f2bf(W[k * DD + n]);
}

// grid_x f32 -> bf16 copy (gather table becomes L2-resident: 102 MB < 192 MB)
__global__ __launch_bounds__(256) void xcast_kernel(const float* __restrict__ x,
                                                    unsigned short* __restrict__ xb, int n4) {
    int i = blockIdx.x * 256 + threadIdx.x;
    if (i < n4) {
        float4 v = ((const float4*)x)[i];
        unsigned p0 = pack2bf(v.x, v.y);
        unsigned p1 = pack2bf(v.z, v.w);
        ((uint2*)xb)[i] = make_uint2(p0, p1);
    }
}

__global__ __launch_bounds__(256) void zero_kernel(float* __restrict__ p, int n) {
    int i = blockIdx.x * 256 + threadIdx.x;
    if (i < n) p[i] = 0.0f;
}

// ---------------------------------------------------------------------------
// Grid path: out = x + MLP_g(x)
// ---------------------------------------------------------------------------
__global__ __launch_bounds__(256) void grid_mlp_kernel(const float* __restrict__ x,
                                                       const unsigned short* __restrict__ xb,
                                                       const unsigned short* __restrict__ W1t,
                                                       const float* __restrict__ b1,
                                                       const unsigned short* __restrict__ W2t,
                                                       const float* __restrict__ b2,
                                                       float* __restrict__ out, int nrows) {
    __shared__ __align__(16) unsigned short xbf[ROWS * DD];
    __shared__ __align__(16) unsigned short hid[ROWS * DD];
    int t    = threadIdx.x;
    int lane = t & 31, wave = t >> 5;
    int row0 = blockIdx.x * ROWS;

    // async-stage 128 bf16 rows: wave w copies rows w, w+8, ...; 512 B per row.
    for (int i = wave; i < ROWS; i += 8) {
        int r  = row0 + i;
        int rc = r < nrows ? r : (nrows - 1);
        unsigned goff = (unsigned)rc * (DD * 2u) + (unsigned)lane * 16u;
        unsigned loff = (unsigned)(size_t)(&xbf[i * DD]) + (unsigned)lane * 16u;
        async_row16(loff, goff, xb);
    }
    async_wait0();
    __syncthreads();

    layer1(xbf, hid, W1t, b1, lane, wave);
    __syncthreads();

    int n0 = wave * 32;
#pragma unroll
    for (int ct = 0; ct < 2; ++ct) {
        int col0 = n0 + ct * 16;
        for (int rt = 0; rt < NRT; ++rt) {
            v8f acc = tile_acc(hid, W2t, b2, rt, col0, lane);
            int mbase = (lane < 16) ? 0 : 8;
            int col   = col0 + (lane & 15);
#pragma unroll
            for (int r = 0; r < 8; ++r) {
                int row = row0 + rt * 16 + mbase + r;
                if (row < nrows) {
                    size_t idx = (size_t)row * DD + col;
                    out[idx] = x[idx] + acc[r];   // residual in f32
                }
            }
        }
    }
}

// ---------------------------------------------------------------------------
// Edge path: agg[edge_dst] += MLP_e(x[edge_src])  (atomic scatter-sum)
// ---------------------------------------------------------------------------
__global__ __launch_bounds__(256) void edge_mlp_kernel(const unsigned short* __restrict__ xb,
                                                       const int* __restrict__ edge_src,
                                                       const int* __restrict__ edge_dst,
                                                       const unsigned short* __restrict__ W1t,
                                                       const float* __restrict__ b1,
                                                       const unsigned short* __restrict__ W2t,
                                                       const float* __restrict__ b2,
                                                       float* __restrict__ agg, int nedges) {
    __shared__ __align__(16) unsigned short xbf[ROWS * DD];
    __shared__ __align__(16) unsigned short hid[ROWS * DD];
    int t    = threadIdx.x;
    int lane = t & 31, wave = t >> 5;
    int e0   = blockIdx.x * ROWS;

    // async gather: row i <- xb[edge_src[e0+i]] (bf16, L2-resident table)
    for (int i = wave; i < ROWS; i += 8) {
        int e  = e0 + i;
        int ec = e < nedges ? e : (nedges - 1);
        int src = edge_src[ec];                       // uniform -> scalar load
        unsigned goff = (unsigned)src * (DD * 2u) + (unsigned)lane * 16u;
        unsigned loff = (unsigned)(size_t)(&xbf[i * DD]) + (unsigned)lane * 16u;
        async_row16(loff, goff, xb);
    }
    async_wait0();
    __syncthreads();

    layer1(xbf, hid, W1t, b1, lane, wave);
    __syncthreads();

    int n0 = wave * 32;
#pragma unroll
    for (int ct = 0; ct < 2; ++ct) {
        int col0 = n0 + ct * 16;
        for (int rt = 0; rt < NRT; ++rt) {
            v8f acc = tile_acc(hid, W2t, b2, rt, col0, lane);
            int mbase = (lane < 16) ? 0 : 8;
            int col   = col0 + (lane & 15);
#pragma unroll
            for (int r = 0; r < 8; ++r) {
                int e = e0 + rt * 16 + mbase + r;
                if (e < nedges) {
                    int dst = edge_dst[e];
                    atomicAdd(&agg[(size_t)dst * DD + col], acc[r]);
                }
            }
        }
    }
}

// ---------------------------------------------------------------------------
// Mesh path: mesh_new = MLP_m(agg)   (tiny: 10k rows)
// ---------------------------------------------------------------------------
__global__ __launch_bounds__(256) void mesh_mlp_kernel(const float* __restrict__ agg,
                                                       const unsigned short* __restrict__ W1t,
                                                       const float* __restrict__ b1,
                                                       const unsigned short* __restrict__ W2t,
                                                       const float* __restrict__ b2,
                                                       float* __restrict__ out, int nrows) {
    __shared__ __align__(16) unsigned short xbf[ROWS * DD];
    __shared__ __align__(16) unsigned short hid[ROWS * DD];
    int t    = threadIdx.x;
    int row0 = blockIdx.x * ROWS;

    // vectorized f32 -> bf16 staging: float4 per thread per iteration
#pragma unroll
    for (int c = 0; c < (ROWS * DD / 4) / 256; ++c) {
        int idx = c * 256 + t;              // float4 index within tile
        int row = idx >> 6;                 // 64 float4 per row
        float4 v;
        if (row0 + row < nrows) v = *(const float4*)(agg + (size_t)row0 * DD + (size_t)idx * 4);
        else                    v = make_float4(0.f, 0.f, 0.f, 0.f);
        unsigned p0 = pack2bf(v.x, v.y);
        unsigned p1 = pack2bf(v.z, v.w);
        *(unsigned*)(&xbf[idx * 4])     = p0;
        *(unsigned*)(&xbf[idx * 4 + 2]) = p1;
    }
    __syncthreads();

    int lane = t & 31, wave = t >> 5;
    layer1(xbf, hid, W1t, b1, lane, wave);
    __syncthreads();

    int n0 = wave * 32;
#pragma unroll
    for (int ct = 0; ct < 2; ++ct) {
        int col0 = n0 + ct * 16;
        for (int rt = 0; rt < NRT; ++rt) {
            v8f acc = tile_acc(hid, W2t, b2, rt, col0, lane);
            int mbase = (lane < 16) ? 0 : 8;
            int col   = col0 + (lane & 15);
#pragma unroll
            for (int r = 0; r < 8; ++r) {
                int row = row0 + rt * 16 + mbase + r;
                if (row < nrows) out[(size_t)row * DD + col] = acc[r];
            }
        }
    }
}

// ---------------------------------------------------------------------------
// Host launcher
// ---------------------------------------------------------------------------
extern "C" void kernel_launch(void* const* d_in, const int* in_sizes, int n_in,
                              void* d_out, int out_size, void* d_ws, size_t ws_size,
                              hipStream_t stream) {
    const float* grid_x   = (const float*)d_in[0];
    const int*   edge_src = (const int*)d_in[1];
    const int*   edge_dst = (const int*)d_in[2];
    // d_in[3] = n_mesh scalar (device); derive sizes host-side instead.
    const float* We1 = (const float*)d_in[4];
    const float* be1 = (const float*)d_in[5];
    const float* We2 = (const float*)d_in[6];
    const float* be2 = (const float*)d_in[7];
    const float* Wm1 = (const float*)d_in[8];
    const float* bm1 = (const float*)d_in[9];
    const float* Wm2 = (const float*)d_in[10];
    const float* bm2 = (const float*)d_in[11];
    const float* Wg1 = (const float*)d_in[12];
    const float* bg1 = (const float*)d_in[13];
    const float* Wg2 = (const float*)d_in[14];
    const float* bg2 = (const float*)d_in[15];

    int n_grid = in_sizes[0] / DD;
    int n_edge = in_sizes[1];
    int n_mesh = out_size / DD - n_grid;

    // Workspace: 6 bf16 transposed weight mats (128 KB each) | agg f32 | x bf16.
    char* ws = (char*)d_ws;
    unsigned short* We1t = (unsigned short*)(ws + 0 * 131072);
    unsigned short* We2t = (unsigned short*)(ws + 1 * 131072);
    unsigned short* Wm1t = (unsigned short*)(ws + 2 * 131072);
    unsigned short* Wm2t = (unsigned short*)(ws + 3 * 131072);
    unsigned short* Wg1t = (unsigned short*)(ws + 4 * 131072);
    unsigned short* Wg2t = (unsigned short*)(ws + 5 * 131072);
    size_t agg_bytes = (size_t)n_mesh * DD * sizeof(float);
    float*          agg = (float*)(ws + 6 * 131072);
    unsigned short* xb  = (unsigned short*)(ws + 6 * 131072 + agg_bytes);

    float* out_grid = (float*)d_out;
    float* out_mesh = (float*)d_out + (size_t)n_grid * DD;

    // 1. weight conversion (f32 [K][N] -> bf16 [N][K])
    transpose_bf16_kernel<<<256, 256, 0, stream>>>(We1, We1t);
    transpose_bf16_kernel<<<256, 256, 0, stream>>>(We2, We2t);
    transpose_bf16_kernel<<<256, 256, 0, stream>>>(Wm1, Wm1t);
    transpose_bf16_kernel<<<256, 256, 0, stream>>>(Wm2, Wm2t);
    transpose_bf16_kernel<<<256, 256, 0, stream>>>(Wg1, Wg1t);
    transpose_bf16_kernel<<<256, 256, 0, stream>>>(Wg2, Wg2t);

    // 2. bf16 cast of grid_x (4 elems/thread)
    int n4 = (n_grid * DD) / 4;
    xcast_kernel<<<(n4 + 255) / 256, 256, 0, stream>>>(grid_x, xb, n4);

    // 3. zero the scatter accumulator (deterministic each call)
    int agg_elems = n_mesh * DD;
    zero_kernel<<<(agg_elems + 255) / 256, 256, 0, stream>>>(agg, agg_elems);

    // 4. grid residual MLP
    grid_mlp_kernel<<<(n_grid + ROWS - 1) / ROWS, 256, 0, stream>>>(
        grid_x, xb, Wg1t, bg1, Wg2t, bg2, out_grid, n_grid);

    // 5. edge MLP + scatter-sum (gather table L2-resident)
    edge_mlp_kernel<<<(n_edge + ROWS - 1) / ROWS, 256, 0, stream>>>(
        xb, edge_src, edge_dst, We1t, be1, We2t, be2, agg, n_edge);

    // 6. mesh MLP
    mesh_mlp_kernel<<<(n_mesh + ROWS - 1) / ROWS, 256, 0, stream>>>(
        agg, Wm1t, bm1, Wm2t, bm2, out_mesh, n_mesh);
}
